// CableLinkModel_2886218023051
// MI455X (gfx1250) — matrix operand order repaired and checked by reading, not compiled, not run on hardware
//
#include <hip/hip_runtime.h>

// ---------------------------------------------------------------------------
// CableLinkModel on gfx1250 (MI455X). All GEMM-shaped work (convs via
// implicit im2col, QKV, attention, FFN, projections) runs through
// v_wmma_f32_16x16x32_bf16 (bf16 in, f32 accum). Everything else is VALU.
// Conv geometry is templated so index math compiles to shifts; LDS tiles are
// staged as packed bf16-pair dwords in fragment order so each WMMA operand is
// two ds_load_b128; GUARD=0 specializations drop all bounds checks for
// tile-aligned problems (no exec-mask churn in the hot loop).
// ---------------------------------------------------------------------------

typedef __attribute__((ext_vector_type(16))) __bf16 v16bf;
typedef __attribute__((ext_vector_type(2)))  __bf16 v2bf;
typedef __attribute__((ext_vector_type(8)))  float v8f;

struct U8frag { uint4 lo, hi; };   // 32B = one v16bf operand

__device__ __forceinline__ unsigned int pack_bf2(float f0, float f1) {
#if __has_builtin(__builtin_amdgcn_cvt_pk_bf16_f32)
  v2bf r = __builtin_amdgcn_cvt_pk_bf16_f32(f0, f1);
  return __builtin_bit_cast(unsigned int, r);
#else
  unsigned int u0 = __builtin_bit_cast(unsigned int, f0);
  u0 += 0x7FFFu + ((u0 >> 16) & 1u);           // round-to-nearest-even
  unsigned int u1 = __builtin_bit_cast(unsigned int, f1);
  u1 += 0x7FFFu + ((u1 >> 16) & 1u);
  return (u0 >> 16) | (u1 & 0xFFFF0000u);
#endif
}

// ---------------------------------------------------------------------------
// WMMA GEMM: C[M,N] = epilogue(alpha * A[M,K] * B[K,N])
//   CONV=0: A fp32 row-major (lda).  CONV=1: implicit im2col from NHWC input
//   (geometry in template params; K index = (ky*KW+kx)*CIN + c).
//   GUARD=0: M%64==0 && N%64==0 && K%32==0 -> no bounds checks anywhere
//   (conv keeps only the padding checks). GUARD=1: fully guarded.
//   B: fp32, element (k,n) at B[k*bsk + n*bsn] (NT via stride swap).
//   epilogue: v = acc*alpha; v *= scale[n]; v += bias[n]; relu.
//   blockIdx.z batching via element strides ab/bb/cb.
// Block: 128 threads (4 waves). Tile 64x64, BK=32.
// ---------------------------------------------------------------------------
template<int CONV, int GUARD, int CIN, int HH, int WW, int KHt, int KWt,
         int ST, int PD, int OHt, int OWt>
__global__ __launch_bounds__(128)
void wmma_gemm_t(const float* __restrict__ A, const float* __restrict__ B,
                 float* __restrict__ C,
                 const float* __restrict__ scale, const float* __restrict__ bias,
                 int relu, float alpha,
                 int M, int N, int K,
                 int lda, int bsk, int bsn, int ldc,
                 long long ab, long long bb, long long cb)
{
  // packed bf16 pairs, fragment-ordered; row stride 20 dwords keeps the
  // 0/16/32/48-byte fragment offsets 16B-aligned and banks well spread.
  __shared__ __align__(16) unsigned int As32[64][20];
  __shared__ __align__(16) unsigned int Bs32[64][20];

  A += (long long)blockIdx.z * ab;
  B += (long long)blockIdx.z * bb;
  C += (long long)blockIdx.z * cb;

  const int m0 = blockIdx.y * 64;
  const int n0 = blockIdx.x * 64;
  const int tid = threadIdx.x;
  const int lane = tid & 31;
  const int wv = tid >> 5;

  // --- staging geometry (fixed per thread across the K loop) ---
  const int akp = tid & 15;        // A k-pair column (k = k0 + 2*akp, +1)
  const int ar0 = tid >> 4;        // A base row; rows = ar0 + 8*i, i<8
  const int bn  = tid & 63;        // B column
  const int bkp0 = tid >> 6;       // B k-pair base; pairs = bkp0 + 2*i, i<8
  const long long bno = (long long)(n0 + bn) * bsn;
  const bool bn_ok = !GUARD || ((n0 + bn) < N);

  // hoisted im2col row decode (compile-time divisors -> shifts)
  int cn[8], cy[8], cx[8];
  if (CONV) {
    #pragma unroll
    for (int i = 0; i < 8; ++i) {
      int m = m0 + ar0 + i * 8;
      int n  = m / (OHt * OWt);
      int rm = m % (OHt * OWt);
      cn[i] = n;
      cy[i] = (rm / OWt) * ST - PD;
      cx[i] = (rm % OWt) * ST - PD;
    }
  }

  v8f acc[4];
  #pragma unroll
  for (int j = 0; j < 4; ++j) acc[j] = (v8f)0.0f;

  for (int k0 = 0; k0 < K; k0 += 32) {
    const int ka = k0 + 2 * akp;
    // ---- stage A tile (64x32 -> 64x16 packed pairs) ----
    if (CONV) {
      int c0, ky0, kx0, c1, ky1, kx1;
      { int k = ka;     c0 = k % CIN; int t = k / CIN; kx0 = t % KWt; ky0 = t / KWt; }
      { int k = ka + 1; c1 = k % CIN; int t = k / CIN; kx1 = t % KWt; ky1 = t / KWt; }
      #pragma unroll
      for (int i = 0; i < 8; ++i) {
        int row = ar0 + i * 8;
        float v0 = 0.0f, v1 = 0.0f;
        bool mok = !GUARD || (m0 + row < M);
        int iy0 = cy[i] + ky0, ix0 = cx[i] + kx0;
        if (mok && (!GUARD || ka < K) &&
            iy0 >= 0 && iy0 < HH && ix0 >= 0 && ix0 < WW)
          v0 = A[(((long long)cn[i] * HH + iy0) * WW + ix0) * CIN + c0];
        int iy1 = cy[i] + ky1, ix1 = cx[i] + kx1;
        if (mok && (!GUARD || ka + 1 < K) &&
            iy1 >= 0 && iy1 < HH && ix1 >= 0 && ix1 < WW)
          v1 = A[(((long long)cn[i] * HH + iy1) * WW + ix1) * CIN + c1];
        As32[row][akp] = pack_bf2(v0, v1);
      }
    } else if (GUARD) {
      #pragma unroll
      for (int i = 0; i < 8; ++i) {
        int row = ar0 + i * 8;
        int m = m0 + row;
        float v0 = (m < M && ka < K)     ? A[(long long)m * lda + ka]     : 0.0f;
        float v1 = (m < M && ka + 1 < K) ? A[(long long)m * lda + ka + 1] : 0.0f;
        As32[row][akp] = pack_bf2(v0, v1);
      }
    } else {
      if (k0 + 32 < K)   // prefetch next A tile
        __builtin_prefetch(&A[(long long)(m0 + ar0) * lda + k0 + 32], 0, 1);
      #pragma unroll
      for (int i = 0; i < 8; ++i) {
        int row = ar0 + i * 8;
        float2 av = *(const float2*)&A[(long long)(m0 + row) * lda + ka];
        As32[row][akp] = pack_bf2(av.x, av.y);
      }
    }
    // ---- stage B tile (32x64 -> 64 cols x 16 packed pairs) ----
    if (GUARD) {
      #pragma unroll
      for (int i = 0; i < 8; ++i) {
        int kp = bkp0 + 2 * i;
        int k = k0 + 2 * kp;
        float v0 = (bn_ok && k < K)     ? B[(long long)k * bsk + bno] : 0.0f;
        float v1 = (bn_ok && k + 1 < K) ? B[(long long)(k + 1) * bsk + bno] : 0.0f;
        Bs32[bn][kp] = pack_bf2(v0, v1);
      }
    } else {
      if (k0 + 32 < K)   // prefetch next B tile
        __builtin_prefetch(&B[(long long)(k0 + 32) * bsk + bno], 0, 1);
      #pragma unroll
      for (int i = 0; i < 8; ++i) {
        int kp = bkp0 + 2 * i;
        long long k = k0 + 2 * kp;
        float v0 = B[k * bsk + bno];
        float v1 = B[(k + 1) * bsk + bno];
        Bs32[bn][kp] = pack_bf2(v0, v1);
      }
    }
    __syncthreads();

    // ---- fragments: two ds_load_b128 each, bit-cast to v16bf ----
    const int mr = (wv << 4) + (lane & 15);
    const int kbpA = (lane < 16) ? 0 : 4;
    U8frag a8;
    a8.lo = *(const uint4*)&As32[mr][kbpA];
    a8.hi = *(const uint4*)&As32[mr][8 + kbpA];
    v16bf af = __builtin_bit_cast(v16bf, a8);

    v16bf bfrag[4];
    #pragma unroll
    for (int j = 0; j < 4; ++j) {
      const int nn = (j << 4) + (lane & 15);
      const int kbpB = (lane < 16) ? 0 : 4;
      U8frag b8;
      b8.lo = *(const uint4*)&Bs32[nn][kbpB];
      b8.hi = *(const uint4*)&Bs32[nn][8 + kbpB];
      bfrag[j] = __builtin_bit_cast(v16bf, b8);
    }
    #pragma unroll
    for (int j = 0; j < 4; ++j)
      acc[j] = __builtin_amdgcn_wmma_f32_16x16x32_bf16(
          false, af, false, bfrag[j], (short)0, acc[j], false, false);
    __syncthreads();
  }

  // ---- epilogue ----
  #pragma unroll
  for (int j = 0; j < 4; ++j) {
    #pragma unroll
    for (int r = 0; r < 8; ++r) {
      int row = m0 + (wv << 4) + r + ((lane < 16) ? 0 : 8);
      int col = n0 + (j << 4) + (lane & 15);
      if (!GUARD || (row < M && col < N)) {
        float v = acc[j][r] * alpha;
        if (scale) v *= scale[col];
        if (bias)  v += bias[col];
        if (relu)  v = fmaxf(v, 0.0f);
        C[(long long)row * ldc + col] = v;
      }
    }
  }
}

// ---------------------------------------------------------------------------
// Elementwise / reduction kernels
// ---------------------------------------------------------------------------
__global__ void nchw_to_nhwc(const float* __restrict__ in, float* __restrict__ out,
                             int N, int C, int H, int W) {
  int idx = blockIdx.x * 256 + threadIdx.x;
  int tot = N * C * H * W;
  if (idx >= tot) return;
  int x = idx % W; int t = idx / W;
  int y = t % H;   t /= H;
  int c = t % C;   int n = t / C;
  out[(((long long)n * H + y) * W + x) * C + c] = in[idx];
}

__global__ void bn_prep(const float* g, const float* b, const float* m,
                        const float* v, float* s, float* t, int C) {
  int c = threadIdx.x;
  if (c < C) {
    float sc = g[c] * rsqrtf(v[c] + 1e-5f);
    s[c] = sc;
    t[c] = b[c] - m[c] * sc;
  }
}

__global__ void pack_w(const float* __restrict__ w, float* __restrict__ bp,
                       int O, int Cin, int KH, int KW, int Kpad) {
  int idx = blockIdx.x * 256 + threadIdx.x;
  if (idx >= Kpad * O) return;
  int k = idx / O, o = idx % O;
  float val = 0.0f;
  int Kt = Cin * KH * KW;
  if (k < Kt) {
    int c = k % Cin; int t = k / Cin;
    int kx = t % KW; int ky = t / KW;
    val = w[(((long long)o * Cin + c) * KH + ky) * KW + kx];
  }
  bp[idx] = val;
}

__global__ void maxpool3s2(const float* __restrict__ in, float* __restrict__ out,
                           int N, int Hin, int Win, int C, int Hout, int Wout) {
  int idx = blockIdx.x * 256 + threadIdx.x;
  int tot = N * Hout * Wout * C;
  if (idx >= tot) return;
  int c = idx % C; int t = idx / C;
  int ox = t % Wout; t /= Wout;
  int oy = t % Hout; int n = t / Hout;
  float m = -1e30f;
  for (int dy = 0; dy < 3; ++dy)
    for (int dx = 0; dx < 3; ++dx) {
      int iy = oy * 2 - 1 + dy, ix = ox * 2 - 1 + dx;
      if (iy >= 0 && iy < Hin && ix >= 0 && ix < Win)
        m = fmaxf(m, in[(((long long)n * Hin + iy) * Win + ix) * C + c]);
    }
  out[idx] = m;
}

__global__ void avgpool_all(const float* __restrict__ in, float* __restrict__ out,
                            int N, int H, int W, int C) {
  int idx = blockIdx.x * 256 + threadIdx.x;
  if (idx >= N * C) return;
  int n = idx / C, c = idx % C;
  float s = 0.0f;
  for (int i = 0; i < H * W; ++i)
    s += in[((long long)n * H * W + i) * C + c];
  out[idx] = s / (float)(H * W);
}

__global__ void add_relu(const float* __restrict__ a, const float* __restrict__ b,
                         float* __restrict__ y, int n) {
  int idx = blockIdx.x * 256 + threadIdx.x;
  if (idx < n) y[idx] = fmaxf(a[idx] + b[idx], 0.0f);
}

__global__ void embed_gather(const float* __restrict__ table, const int* __restrict__ lab,
                             float* __restrict__ out, int n, int dim) {
  int idx = blockIdx.x * 256 + threadIdx.x;
  if (idx >= n * dim) return;
  int i = idx / dim, d = idx % dim;
  out[idx] = table[(long long)lab[i] * dim + d];
}

__global__ void concat3(const float* __restrict__ a, const float* __restrict__ b,
                        const float* __restrict__ c, float* __restrict__ y,
                        int rows, int d1, int d2, int d3) {
  int dt = d1 + d2 + d3;
  int idx = blockIdx.x * 256 + threadIdx.x;
  if (idx >= rows * dt) return;
  int r = idx / dt, col = idx % dt;
  float v;
  if (col < d1)            v = a[(long long)r * d1 + col];
  else if (col < d1 + d2)  v = b[(long long)r * d2 + (col - d1)];
  else                     v = c[(long long)r * d3 + (col - d1 - d2)];
  y[idx] = v;
}

// LayerNorm over last dim (Dn <= 512), optional fused residual, per-row block.
__global__ __launch_bounds__(256)
void ln_kernel(const float* __restrict__ x, const float* __restrict__ res,
               const float* __restrict__ g, const float* __restrict__ b,
               float* __restrict__ y, int Dn) {
  __shared__ float red[256];
  int row = blockIdx.x, tid = threadIdx.x;
  const float* xr = x + (long long)row * Dn;
  const float* rr = res ? res + (long long)row * Dn : nullptr;
  bool h0 = tid < Dn, h1 = (tid + 256) < Dn;
  float v0 = 0.0f, v1 = 0.0f;
  if (h0) { v0 = xr[tid];       if (rr) v0 += rr[tid]; }
  if (h1) { v1 = xr[tid + 256]; if (rr) v1 += rr[tid + 256]; }
  red[tid] = v0 + v1; __syncthreads();
  for (int s = 128; s > 0; s >>= 1) { if (tid < s) red[tid] += red[tid + s]; __syncthreads(); }
  float mu = red[0] / (float)Dn; __syncthreads();
  float d0 = h0 ? v0 - mu : 0.0f, d1 = h1 ? v1 - mu : 0.0f;
  red[tid] = d0 * d0 + d1 * d1; __syncthreads();
  for (int s = 128; s > 0; s >>= 1) { if (tid < s) red[tid] += red[tid + s]; __syncthreads(); }
  float rs = rsqrtf(red[0] / (float)Dn + 1e-5f);
  float* yr = y + (long long)row * Dn;
  if (h0) yr[tid]       = d0 * rs * g[tid] + b[tid];
  if (h1) yr[tid + 256] = d1 * rs * g[tid + 256] + b[tid + 256];
}

// Softmax over rows of length 256 with key-padding mask. Row = (b*8+h)*256+i.
__global__ __launch_bounds__(256)
void softmax_mask(float* __restrict__ S, const unsigned char* __restrict__ mask) {
  __shared__ float red[256];
  int row = blockIdx.x, j = threadIdx.x;
  int b = row >> 11;                       // 8 heads * 256 rows per batch
  float x = S[(long long)row * 256 + j];
  if (mask[b * 256 + j] == 0) x = -1e9f;
  red[j] = x; __syncthreads();
  for (int s = 128; s > 0; s >>= 1) { if (j < s) red[j] = fmaxf(red[j], red[j + s]); __syncthreads(); }
  float mx = red[0]; __syncthreads();
  float e = __expf(x - mx);
  red[j] = e; __syncthreads();
  for (int s = 128; s > 0; s >>= 1) { if (j < s) red[j] += red[j + s]; __syncthreads(); }
  S[(long long)row * 256 + j] = e / red[0];
}

// One wave per (b,i,j) pair: t = relu(hi[b,i]+hj[b,j]+bl); LN(128); dot w2 + b2.
__global__ __launch_bounds__(256)
void pair_kernel(const float* __restrict__ hi, const float* __restrict__ hj,
                 const float* __restrict__ bl, const float* __restrict__ g,
                 const float* __restrict__ be, const float* __restrict__ w2,
                 const float* __restrict__ b2, float* __restrict__ L) {
  int wv = threadIdx.x >> 5, lane = threadIdx.x & 31;
  int p = blockIdx.x * 8 + wv;
  if (p >= 2 * 256 * 256) return;
  int b = p >> 16, rem = p & 65535, i = rem >> 8, j = rem & 255;
  float4 a4  = ((const float4*)(hi + ((long long)(b * 256 + i)) * 128))[lane];
  float4 c4  = ((const float4*)(hj + ((long long)(b * 256 + j)) * 128))[lane];
  float4 bl4 = ((const float4*)bl)[lane];
  float t0 = fmaxf(a4.x + c4.x + bl4.x, 0.0f);
  float t1 = fmaxf(a4.y + c4.y + bl4.y, 0.0f);
  float t2 = fmaxf(a4.z + c4.z + bl4.z, 0.0f);
  float t3 = fmaxf(a4.w + c4.w + bl4.w, 0.0f);
  float s = t0 + t1 + t2 + t3;
  for (int m = 16; m >= 1; m >>= 1) s += __shfl_xor(s, m, 32);
  float mu = s * (1.0f / 128.0f);
  float d0 = t0 - mu, d1 = t1 - mu, d2 = t2 - mu, d3 = t3 - mu;
  float vs = d0 * d0 + d1 * d1 + d2 * d2 + d3 * d3;
  for (int m = 16; m >= 1; m >>= 1) vs += __shfl_xor(vs, m, 32);
  float rs = rsqrtf(vs * (1.0f / 128.0f) + 1e-5f);
  float4 g4 = ((const float4*)g)[lane];
  float4 e4 = ((const float4*)be)[lane];
  float4 w4 = ((const float4*)w2)[lane];
  float dot = (d0 * rs * g4.x + e4.x) * w4.x
            + (d1 * rs * g4.y + e4.y) * w4.y
            + (d2 * rs * g4.z + e4.z) * w4.z
            + (d3 * rs * g4.w + e4.w) * w4.w;
  for (int m = 16; m >= 1; m >>= 1) dot += __shfl_xor(dot, m, 32);
  if (lane == 0) L[p] = dot + b2[0];
}

__global__ void sym_mask(const float* __restrict__ L, const unsigned char* __restrict__ mask,
                         float* __restrict__ out) {
  int idx = blockIdx.x * 256 + threadIdx.x;
  if (idx >= 2 * 256 * 256) return;
  int b = idx >> 16, rem = idx & 65535, i = rem >> 8, j = rem & 255;
  bool mm = mask[b * 256 + i] && mask[b * 256 + j];
  float v = 0.5f * (L[idx] + L[(b << 16) | (j << 8) | i]);
  out[idx] = mm ? v : -1e9f;
}

// ---------------------------------------------------------------------------
// Host-side helpers
// ---------------------------------------------------------------------------
static void run_gemm(hipStream_t st, const float* A, const float* B, float* C,
                     const float* scale, const float* bias, int relu, float alpha,
                     int M, int N, int K, int lda, int bsk, int bsn, int ldc,
                     long long ab, long long bb, long long cb, int nb) {
  dim3 grid((N + 63) / 64, (M + 63) / 64, nb);
  bool aligned = (M % 64 == 0) && (N % 64 == 0) && (K % 32 == 0);
  if (aligned)
    wmma_gemm_t<0, 0, 1, 1, 1, 1, 1, 1, 0, 1, 1><<<grid, 128, 0, st>>>(
        A, B, C, scale, bias, relu, alpha, M, N, K, lda, bsk, bsn, ldc, ab, bb, cb);
  else
    wmma_gemm_t<0, 1, 1, 1, 1, 1, 1, 1, 0, 1, 1><<<grid, 128, 0, st>>>(
        A, B, C, scale, bias, relu, alpha, M, N, K, lda, bsk, bsn, ldc, ab, bb, cb);
}

template<int CIN, int HH, int WW, int KH, int KW, int ST, int PD>
static void run_conv_t(hipStream_t st, const float* in, const float* w,
                       const float* g, const float* bb_, const float* m_,
                       const float* v_, float* out, float* wpack, float* sbuf,
                       float* tbuf, int O, int relu) {
  constexpr int OH = (HH + 2 * PD - KH) / ST + 1;
  constexpr int OW = (WW + 2 * PD - KW) / ST + 1;
  constexpr int GUARD = (CIN * KH * KW % 32) ? 1 : 0;  // M,N always 64-aligned
  int Kt = CIN * KH * KW;
  int Kpad = (Kt + 31) & ~31;
  pack_w<<<(Kpad * O + 255) / 256, 256, 0, st>>>(w, wpack, O, CIN, KH, KW, Kpad);
  bn_prep<<<1, O, 0, st>>>(g, bb_, m_, v_, sbuf, tbuf, O);
  int M = 512 * OH * OW;
  dim3 grid((O + 63) / 64, (M + 63) / 64, 1);
  wmma_gemm_t<1, GUARD, CIN, HH, WW, KH, KW, ST, PD, OH, OW><<<grid, 128, 0, st>>>(
      in, wpack, out, sbuf, tbuf, relu, 1.0f, M, O, Kt, 0, O, 1, O, 0, 0, 0);
}

// ---------------------------------------------------------------------------
// kernel_launch
//
// Param flattening assumption (depth-first, dict insertion order):
//  0 crops  1 pos  2 labels(int32)  3 mask(bool/u8)
//  4 conv1_w ; 5..8 bn1(g,b,m,v)
//  layer1[0]: 9 c1, 10..13 bn1, 14 c2, 15..18 bn2
//  layer1[1]: 19 c1, 20..23 bn1, 24 c2, 25..28 bn2
//  layer2[0]: 29 c1, 30..33 bn1, 34 c2, 35..38 bn2, 39 down, 40..43 dbn
//  layer2[1]: 44 c1, 45..48 bn1, 49 c2, 50..53 bn2
//  54 visW 55 visb ; 56 embed ; 57 pos1W 58 pos1b 59 pos2W 60 pos2b
//  61 fuseW 62 fuseb ; 63 fuse_ln_g 64 fuse_ln_b
//  tr layer l at 65+16*l: Wq,bq,Wk,bk,Wv,bv,Wo,bo,ln1g,ln1b,W1,b1,W2,b2,ln2g,ln2b
//  113 lp1W 114 lp1b 115 lp_ln_g 116 lp_ln_b 117 lp2W 118 lp2b
// ---------------------------------------------------------------------------
extern "C" void kernel_launch(void* const* d_in, const int* in_sizes, int n_in,
                              void* d_out, int out_size, void* d_ws, size_t ws_size,
                              hipStream_t stream) {
  (void)in_sizes; (void)n_in; (void)out_size; (void)ws_size;
  auto F = [&](int i) { return (const float*)d_in[i]; };
  const float* crops = F(0);
  const float* pos   = F(1);
  const int*   labels = (const int*)d_in[2];
  const unsigned char* mask = (const unsigned char*)d_in[3];
  float* out = (float*)d_out;

  // ---- workspace carve-up ----
  float* ws = (float*)d_ws;
  size_t off = 0;
  auto alloc = [&](size_t n) { float* p = ws + off; off += (n + 63) & ~(size_t)63; return p; };
  float* cnhwc = alloc(512ll * 32 * 32 * 3);   // crops NHWC
  float* x1 = alloc(512ll * 16 * 16 * 64);     // conv1 out
  float* x2 = alloc(512ll * 8 * 8 * 64);       // running feature @64ch
  float* x3 = alloc(512ll * 8 * 8 * 64);
  float* x4 = alloc(512ll * 8 * 8 * 64);
  float* y1 = alloc(512ll * 4 * 4 * 128);      // running feature @128ch
  float* y2 = alloc(512ll * 4 * 4 * 128);
  float* y3 = alloc(512ll * 4 * 4 * 128);
  float* pooled = alloc(512 * 128);
  float* vis = alloc(512 * 128);
  float* lbl = alloc(512 * 64);
  float* pf1 = alloc(512 * 16);
  float* pf  = alloc(512 * 32);
  float* fin = alloc(512 * 224);
  float* ctx = alloc(512 * 256);
  float* q   = alloc(512 * 256);
  float* kk  = alloc(512 * 256);
  float* vvb = alloc(512 * 256);
  float* sc  = alloc(16ll * 256 * 256);        // scores (2,8,256,256)
  float* ao  = alloc(512 * 256);               // attention out
  float* t2  = alloc(512 * 256);               // generic (512,256) temp
  float* mid = alloc(512 * 512);               // FFN hidden
  float* hi  = alloc(512 * 128);
  float* hj  = alloc(512 * 128);
  float* lgt = alloc(2ll * 256 * 256);
  float* wpack = alloc(1152 * 128);            // largest packed conv weight
  float* sbuf = alloc(128);
  float* tbuf = alloc(128);

  // ---- visual encoder ----
  nchw_to_nhwc<<<(512 * 3 * 32 * 32 + 255) / 256, 256, 0, stream>>>(crops, cnhwc, 512, 3, 32, 32);
  run_conv_t<3, 32, 32, 7, 7, 2, 3>(stream, cnhwc, F(4), F(5), F(6), F(7), F(8),
                                    x1, wpack, sbuf, tbuf, 64, 1);
  maxpool3s2<<<(512 * 8 * 8 * 64 + 255) / 256, 256, 0, stream>>>(x1, x2, 512, 16, 16, 64, 8, 8);

  // layer1: two BasicBlocks 64->64, 8x8
  for (int blk = 0; blk < 2; ++blk) {
    int bi = 9 + blk * 10;
    run_conv_t<64, 8, 8, 3, 3, 1, 1>(stream, x2, F(bi), F(bi + 1), F(bi + 2),
                                     F(bi + 3), F(bi + 4), x3, wpack, sbuf, tbuf, 64, 1);
    run_conv_t<64, 8, 8, 3, 3, 1, 1>(stream, x3, F(bi + 5), F(bi + 6), F(bi + 7),
                                     F(bi + 8), F(bi + 9), x4, wpack, sbuf, tbuf, 64, 0);
    add_relu<<<(512 * 8 * 8 * 64 + 255) / 256, 256, 0, stream>>>(x4, x2, x2, 512 * 8 * 8 * 64);
  }

  // layer2 block0: 64->128 stride2 with downsample
  run_conv_t<64, 8, 8, 3, 3, 2, 1>(stream, x2, F(29), F(30), F(31), F(32), F(33),
                                   y2, wpack, sbuf, tbuf, 128, 1);
  run_conv_t<128, 4, 4, 3, 3, 1, 1>(stream, y2, F(34), F(35), F(36), F(37), F(38),
                                    y3, wpack, sbuf, tbuf, 128, 0);
  run_conv_t<64, 8, 8, 1, 1, 2, 0>(stream, x2, F(39), F(40), F(41), F(42), F(43),
                                   y1, wpack, sbuf, tbuf, 128, 0);
  add_relu<<<(512 * 4 * 4 * 128 + 255) / 256, 256, 0, stream>>>(y3, y1, y1, 512 * 4 * 4 * 128);
  // layer2 block1: 128->128
  run_conv_t<128, 4, 4, 3, 3, 1, 1>(stream, y1, F(44), F(45), F(46), F(47), F(48),
                                    y2, wpack, sbuf, tbuf, 128, 1);
  run_conv_t<128, 4, 4, 3, 3, 1, 1>(stream, y2, F(49), F(50), F(51), F(52), F(53),
                                    y3, wpack, sbuf, tbuf, 128, 0);
  add_relu<<<(512 * 4 * 4 * 128 + 255) / 256, 256, 0, stream>>>(y3, y1, y1, 512 * 4 * 4 * 128);

  avgpool_all<<<(512 * 128 + 255) / 256, 256, 0, stream>>>(y1, pooled, 512, 4, 4, 128);
  run_gemm(stream, pooled, F(54), vis, nullptr, F(55), 0, 1.0f,
           512, 128, 128, 128, 128, 1, 128, 0, 0, 0, 1);

  // ---- label + position encoders ----
  embed_gather<<<(512 * 64 + 255) / 256, 256, 0, stream>>>(F(56), labels, lbl, 512, 64);
  run_gemm(stream, pos, F(57), pf1, nullptr, F(58), 1, 1.0f,
           512, 16, 2, 2, 16, 1, 16, 0, 0, 0, 1);
  run_gemm(stream, pf1, F(59), pf, nullptr, F(60), 0, 1.0f,
           512, 32, 16, 16, 32, 1, 32, 0, 0, 0, 1);

  // ---- fusion MLP + LN ----
  concat3<<<(512 * 224 + 255) / 256, 256, 0, stream>>>(vis, lbl, pf, fin, 512, 128, 64, 32);
  run_gemm(stream, fin, F(61), t2, nullptr, F(62), 1, 1.0f,
           512, 256, 224, 224, 256, 1, 256, 0, 0, 0, 1);
  ln_kernel<<<512, 256, 0, stream>>>(t2, nullptr, F(63), F(64), ctx, 256);

  // ---- transformer encoder (3 layers, post-norm) ----
  const float iscale = 0.17677669529663687f; // 1/sqrt(32)
  for (int l = 0; l < 3; ++l) {
    int tb = 65 + 16 * l;
    run_gemm(stream, ctx, F(tb + 0), q,   nullptr, F(tb + 1), 0, 1.0f,
             512, 256, 256, 256, 256, 1, 256, 0, 0, 0, 1);
    run_gemm(stream, ctx, F(tb + 2), kk,  nullptr, F(tb + 3), 0, 1.0f,
             512, 256, 256, 256, 256, 1, 256, 0, 0, 0, 1);
    run_gemm(stream, ctx, F(tb + 4), vvb, nullptr, F(tb + 5), 0, 1.0f,
             512, 256, 256, 256, 256, 1, 256, 0, 0, 0, 1);
    for (int b = 0; b < 2; ++b) {
      // scores[b,h] = Q_bh (256x32) @ K_bh^T (32x256) * 1/sqrt(hd); z = head
      run_gemm(stream, q + b * 65536, kk + b * 65536, sc + (long long)b * 8 * 65536,
               nullptr, nullptr, 0, iscale,
               256, 256, 32, 256, 1, 256, 256, 32, 32, 65536, 8);
    }
    softmax_mask<<<16 * 256, 256, 0, stream>>>(sc, mask);
    for (int b = 0; b < 2; ++b) {
      // attn_out[b,:,h*32:(h+1)*32] = attn[b,h] (256x256) @ V_bh (256x32)
      run_gemm(stream, sc + (long long)b * 8 * 65536, vvb + b * 65536, ao + b * 65536,
               nullptr, nullptr, 0, 1.0f,
               256, 32, 256, 256, 256, 1, 256, 65536, 32, 32, 8);
    }
    run_gemm(stream, ao, F(tb + 6), t2, nullptr, F(tb + 7), 0, 1.0f,
             512, 256, 256, 256, 256, 1, 256, 0, 0, 0, 1);
    ln_kernel<<<512, 256, 0, stream>>>(ctx, t2, F(tb + 8), F(tb + 9), ctx, 256);
    run_gemm(stream, ctx, F(tb + 10), mid, nullptr, F(tb + 11), 1, 1.0f,
             512, 512, 256, 256, 512, 1, 512, 0, 0, 0, 1);
    run_gemm(stream, mid, F(tb + 12), t2, nullptr, F(tb + 13), 0, 1.0f,
             512, 256, 512, 512, 256, 1, 256, 0, 0, 0, 1);
    ln_kernel<<<512, 256, 0, stream>>>(ctx, t2, F(tb + 14), F(tb + 15), ctx, 256);
  }

  // ---- all-pairs link predictor ----
  run_gemm(stream, ctx, F(113), hi, nullptr, nullptr, 0, 1.0f,
           512, 128, 256, 256, 128, 1, 128, 0, 0, 0, 1);
  run_gemm(stream, ctx, F(113) + 256 * 128, hj, nullptr, nullptr, 0, 1.0f,
           512, 128, 256, 256, 128, 1, 128, 0, 0, 0, 1);
  pair_kernel<<<16384, 256, 0, stream>>>(hi, hj, F(114), F(115), F(116),
                                         F(117), F(118), lgt);
  sym_mask<<<512, 256, 0, stream>>>(lgt, mask, out);
}
